// KNNEmbedding_55164559949910
// MI455X (gfx1250) — matrix operand order repaired and checked by reading, not compiled, not run on hardware
//
#include <hip/hip_runtime.h>
#include <hip/hip_bf16.h>
#include <stdint.h>

// ---------------------------------------------------------------------------
// KNN embedding for MI455X (gfx1250, wave32, WMMA + TDM).
//
// Shapes: B=32, N=2048, D=16 (2D=32 channels), K=16 neighbors, DM=512.
// Dominant cost: [65536 x 512] x [512 x 1024] GEMM (~69 GFLOP) -> WMMA f16,
// with B-fragments staged block-wide into LDS by the Tensor Data Mover.
// ---------------------------------------------------------------------------

#define BB   32
#define NN   2048
#define DD   16
#define CH   32          // 2*D channels
#define KNB  16          // neighbors
#define DM   512
#define FDIM 512         // K * 2D

typedef __attribute__((ext_vector_type(16))) _Float16 v16h;
typedef __attribute__((ext_vector_type(8)))  float    v8f;
typedef __attribute__((ext_vector_type(4)))  unsigned u32x4;
typedef __attribute__((ext_vector_type(8)))  int      i32x8;
typedef __attribute__((ext_vector_type(4)))  int      i32x4;

union FragU { v16h v; uint4 q[2]; };

// ---------------------------------------------------------------------------
// K1: masked normalize.  One block per batch, 256 threads.
// ---------------------------------------------------------------------------
__global__ void __launch_bounds__(256) k_normalize(
    const float* __restrict__ x, const float* __restrict__ features,
    const unsigned char* __restrict__ mask,
    float* __restrict__ xn, float* __restrict__ xcrd)
{
    const int b = blockIdx.x;
    const int t = threadIdx.x;
    const int c = t & 31;
    const int r = t >> 5;

    __shared__ float s_sum[CH][8];
    __shared__ float s_sq [CH][8];
    __shared__ float s_cnt[8];
    __shared__ float s_mean[CH];
    __shared__ float s_inv [CH];

    const bool featd = features[b * DD + (c & 15)] > 0.1f;

    float sum = 0.f, sq = 0.f, cnt = 0.f;
    for (int n = r; n < NN; n += 8) {
        const float m  = mask[b * NN + n] ? 1.f : 0.f;
        const float xv = x[((size_t)(b * NN + n)) * DD + (c & 15)];
        const float v  = (c < DD) ? (featd ? 0.f : xv) : (featd ? xv : 0.f);
        sum += v * m;
        sq  += v * v * m;
        cnt += m;
    }
    s_sum[c][r] = sum;
    s_sq [c][r] = sq;
    if (c == 0) s_cnt[r] = cnt;
    __syncthreads();

    if (t < CH) {
        float ts = 0.f, tq = 0.f, tc = 0.f;
        for (int i = 0; i < 8; ++i) { ts += s_sum[t][i]; tq += s_sq[t][i]; tc += s_cnt[i]; }
        tc = fmaxf(tc, 1.0f);
        const float mean = ts / tc;
        float var = tq / tc - mean * mean;
        var = fmaxf(var, 0.f);
        s_mean[t] = mean;
        s_inv[t]  = 1.0f / (sqrtf(var) + 1e-5f);
    }
    __syncthreads();

    const float mean = s_mean[c];
    const float inv  = s_inv[c];
    for (int n = r; n < NN; n += 8) {
        const float xv = x[((size_t)(b * NN + n)) * DD + (c & 15)];
        const float v  = (c < DD) ? (featd ? 0.f : xv) : (featd ? xv : 0.f);
        float z = (v - mean) * inv;
        z = fminf(fmaxf(z, -10.0f), 10.0f);
        xn[((size_t)(b * NN + n)) * CH + c] = z;
        if (c < DD) xcrd[((size_t)(b * NN + n)) * DD + c] = v;   // raw coords
    }
}

// ---------------------------------------------------------------------------
// K2: k-nearest-neighbors.  One wave per query row; 8 waves/block.
// LDS: 8 rows x 2048 d2 values = 64 KB.  16 iterative argmins via packed
// (d2_bits<<32 | j) u64 wave-min -> sorted ascending, lowest-index tiebreak.
// ---------------------------------------------------------------------------
__device__ inline float sq4(float4 a, float4 b) {
    float dx = a.x - b.x, dy = a.y - b.y, dz = a.z - b.z, dw = a.w - b.w;
    return dx * dx + dy * dy + dz * dz + dw * dw;
}

__global__ void __launch_bounds__(256) k_knn(
    const float* __restrict__ xcrd, const unsigned char* __restrict__ mask,
    int* __restrict__ idx)
{
    __shared__ float dist[8][NN];                     // 64 KB
    const int w    = threadIdx.x >> 5;
    const int lane = threadIdx.x & 31;
    const int row  = blockIdx.x * 8 + w;              // global query row
    const int b    = row >> 11;
    const int n    = row & (NN - 1);

    const float* base = xcrd + ((size_t)b * NN) * DD;
    const float4* qp = (const float4*)(base + n * DD);
    const float4 q0 = qp[0], q1 = qp[1], q2 = qp[2], q3 = qp[3];

    for (int j = lane; j < NN; j += 32) {
        const float4* pp = (const float4*)(base + j * DD);
        float d2 = sq4(q0, pp[0]) + sq4(q1, pp[1]) + sq4(q2, pp[2]) + sq4(q3, pp[3]);
        if (!mask[b * NN + j]) d2 = 3.0e38f;
        dist[w][j] = d2;
    }
    __syncthreads();

    for (int it = 0; it < KNB; ++it) {
        unsigned long long best = ~0ull;
        for (int j = lane; j < NN; j += 32) {
            const unsigned db = __float_as_uint(dist[w][j]);   // d2 >= 0: order-preserving
            const unsigned long long key =
                ((unsigned long long)db << 32) | (unsigned)j;
            best = key < best ? key : best;
        }
        for (int s = 16; s > 0; s >>= 1) {
            const unsigned long long o = __shfl_xor(best, s, 32);
            best = o < best ? o : best;
        }
        const int jstar = (int)(unsigned)best;
        if (lane == 0) {
            idx[(size_t)row * KNB + it] = jstar;
            dist[w][jstar] = 3.0e38f;                 // exclude from next round
        }
        __syncthreads();
    }
}

// ---------------------------------------------------------------------------
// K3: pack W[1024][512] (f32) into f16 WMMA B-fragments.
// 16-bit B 32x16 layout: lane l holds col l%16; VGPR g, half e -> element s=2g+e,
// k = 8h+s (s<8) / 16+8h+(s-8) (s>=8), h = l/16.
//   Wh[((tile_o*16+kk)*32 + lane)*8 + g] =
//       pack_f16(W[tile_o*16+(lane&15)][kk*32+k0], W[tile_o*16+(lane&15)][kk*32+k0+1])
// => each 16-output tile_o occupies a contiguous 16 KB block (TDM-friendly).
// ---------------------------------------------------------------------------
__global__ void __launch_bounds__(256) k_packW(
    const float* __restrict__ W, unsigned* __restrict__ Wh)
{
    const int e      = blockIdx.x * 256 + threadIdx.x;   // 0..262143
    const int g      = e & 7;
    const int lane   = (e >> 3) & 31;
    const int tk     = e >> 8;                           // tile_o*16 + kk
    const int kk     = tk & 15;
    const int tile_o = tk >> 4;

    const int kf = kk * 32 + 8 * (lane >> 4) + 2 * g + ((g >= 4) ? 8 : 0); // global f index
    const int oA = tile_o * 16 + (lane & 15);

    union { _Float16 h[2]; unsigned u; } u;
    u.h[0] = (_Float16)W[(size_t)oA * FDIM + kf];
    u.h[1] = (_Float16)W[(size_t)oA * FDIM + kf + 1];
    Wh[e] = u.u;
}

// ---------------------------------------------------------------------------
// TDM: 1-D tile load of `n8` 8-byte units from global `gsrc` to LDS `ldsOff`.
// D# per CDNA5 ISA ch.8: group0 = {count=1 | lds_addr | global_addr | type=2},
// group1 = {data_size=8B, tensor_dim0=tile_dim0=n8, tile_dim1=1, stride=n8},
// groups 2/3 zero (<=2-D tensor).  Tracked by TENSORcnt.
// This toolchain exposes the 6-arg builtin:
//   (uint32x4 g0, int32x8 g1, int32x4 g2, int32x4 g3, int32x8 extra, i32 cpol)
// ---------------------------------------------------------------------------
__device__ __forceinline__ void tdm_load_1d(const void* gsrc, unsigned ldsOff,
                                            unsigned n8)
{
    const unsigned long long ga = (unsigned long long)(uintptr_t)gsrc;
    u32x4 g0;
    g0[0] = 1u;                                              // count=1, user D#
    g0[1] = ldsOff;                                          // lds_addr (bytes)
    g0[2] = (unsigned)ga;                                    // global_addr[31:0]
    g0[3] = (unsigned)((ga >> 32) & 0x01FFFFFFu) | (2u << 30); // [56:32] | type=2

    i32x8 g1;
    g1[0] = (int)(3u << 16);                                 // data_size = 8 bytes
    g1[1] = (int)((n8 & 0xFFFFu) << 16);                     // tensor_dim0[15:0]
    g1[2] = (int)(((n8 >> 16) & 0xFFFFu) | (1u << 16));      // tdim0[31:16]|tdim1=1
    g1[3] = (int)((n8 & 0xFFFFu) << 16);                     // tile_dim0 = n8
    g1[4] = 1;                                               // tile_dim1 = 1
    g1[5] = (int)n8;                                         // tensor_dim0_stride
    g1[6] = 0;
    g1[7] = 0;

    const i32x4 z4 = {};
    const i32x8 z8 = {};
    __builtin_amdgcn_tensor_load_to_lds(g0, g1, z4, z4, z8, 0);
}

// ---------------------------------------------------------------------------
// K4: gather + GEMM + GLU (WMMA + TDM).
// 8 waves/block; each wave owns a 16-query tile, A tile (16x512 f16) built
// directly into 128 VGPRs (A-frag c-set for lane-half h = {8h..8h+7,
// 16+8h..+7} -> two contiguous 8-float segments of xn[j]-xn[m]; kstep ==
// neighbor since FDIM = K*32).
//
// B fragments are shared by all waves: wave 0 TDM-copies the two 16 KB
// fragment tiles (outputs p*16.. and 512+p*16..) for pair p+1 into a
// double-buffered 64 KB LDS region while everyone computes pair p.
// ---------------------------------------------------------------------------
__global__ void __launch_bounds__(256, 1) k_gemm_glu(
    const float* __restrict__ xn, const int* __restrict__ idx,
    const unsigned* __restrict__ Wh, const float* __restrict__ bias,
    float* __restrict__ y)
{
    extern __shared__ char smem[];                       // 2 x 32 KB B buffers
    const int w    = threadIdx.x >> 5;
    const int lane = threadIdx.x & 31;
    const int col  = lane & 15;
    const int hh   = lane >> 4;

    const int qtile = blockIdx.x * 8 + w;
    const int q0    = qtile << 4;                        // flat row base (b*N+n)
    const int b     = q0 >> 11;

    const int nm = q0 + col;                             // this lane's flat A row
    const float* xm_p = xn + (size_t)nm * CH;
    const int c0 = hh * 8;
    const float4 m0 = *(const float4*)(xm_p + c0);
    const float4 m1 = *(const float4*)(xm_p + c0 + 4);
    const float4 m2 = *(const float4*)(xm_p + c0 + 16);
    const float4 m3 = *(const float4*)(xm_p + c0 + 20);

    const int* idxp = idx + (size_t)nm * KNB;

    // ---- build the full 16-kstep A tile directly in registers (16x8 VGPRs) ----
    FragU afr[KNB];
#pragma unroll
    for (int kk = 0; kk < KNB; ++kk) {
        const int j = idxp[kk];
        const float* xj = xn + ((size_t)(b << 11) + j) * CH;
        const float4 a0 = *(const float4*)(xj + c0);
        const float4 a1 = *(const float4*)(xj + c0 + 4);
        const float4 a2 = *(const float4*)(xj + c0 + 16);
        const float4 a3 = *(const float4*)(xj + c0 + 20);

        FragU f;
        f.v[0]  = (_Float16)(a0.x - m0.x);
        f.v[1]  = (_Float16)(a0.y - m0.y);
        f.v[2]  = (_Float16)(a0.z - m0.z);
        f.v[3]  = (_Float16)(a0.w - m0.w);
        f.v[4]  = (_Float16)(a1.x - m1.x);
        f.v[5]  = (_Float16)(a1.y - m1.y);
        f.v[6]  = (_Float16)(a1.z - m1.z);
        f.v[7]  = (_Float16)(a1.w - m1.w);
        f.v[8]  = (_Float16)(a2.x - m2.x);
        f.v[9]  = (_Float16)(a2.y - m2.y);
        f.v[10] = (_Float16)(a2.z - m2.z);
        f.v[11] = (_Float16)(a2.w - m2.w);
        f.v[12] = (_Float16)(a3.x - m3.x);
        f.v[13] = (_Float16)(a3.y - m3.y);
        f.v[14] = (_Float16)(a3.z - m3.z);
        f.v[15] = (_Float16)(a3.w - m3.w);
        afr[kk] = f;
    }

    const char* WhB = (const char*)Wh;                   // tile_o block = 16 KB

    // prime the pipeline: pair 0 into buffer 0 (wave 0 drives the TDM)
    if (w == 0) {
        tdm_load_1d(WhB + (size_t)0 * 16384,  0u,         2048u);  // bA tile
        tdm_load_1d(WhB + (size_t)32 * 16384, 16384u,     2048u);  // bG tile
    }

    // ---- 32 output pairs: (o in [p*16, p*16+16)) and (o+512) ----
#pragma unroll 1
    for (int p = 0; p < 32; ++p) {
        if (w == 0) {
            if (p < 31) {
                const unsigned d = (unsigned)((p + 1) & 1) * 32768u;
                tdm_load_1d(WhB + (size_t)(p + 1)      * 16384, d,          2048u);
                tdm_load_1d(WhB + (size_t)(p + 1 + 32) * 16384, d + 16384u, 2048u);
                __builtin_amdgcn_s_wait_tensorcnt(2);    // pair p landed (in-order)
            } else {
                __builtin_amdgcn_s_wait_tensorcnt(0);    // last pair landed
            }
        }
        __syncthreads();                                 // B(pair p) visible to all

        const char* bufA = smem + (p & 1) * 32768;
        const char* bufG = bufA + 16384;

        v8f accA = {};
        v8f accG = {};
#pragma unroll
        for (int kk = 0; kk < KNB; ++kk) {
            const int off = kk * 1024 + lane * 32;
            FragU ba, bg;
            ba.q[0] = *(const uint4*)(bufA + off);
            ba.q[1] = *(const uint4*)(bufA + off + 16);
            bg.q[0] = *(const uint4*)(bufG + off);
            bg.q[1] = *(const uint4*)(bufG + off + 16);
            accA = __builtin_amdgcn_wmma_f32_16x16x32_f16(
                false, afr[kk].v, false, ba.v, (short)0, accA, false, false);
            accG = __builtin_amdgcn_wmma_f32_16x16x32_f16(
                false, afr[kk].v, false, bg.v, (short)0, accG, false, false);
        }
        __syncthreads();                                 // done reading this buffer

        const float ba_s = bias[p * 16 + col];
        const float bg_s = bias[DM + p * 16 + col];
        // C/D layout: lane&15 = column; VGPR v = row v (lanes 0-15) / v+8 (16-31)
        float* yrow = y + ((size_t)(q0 + hh * 8)) * DM + p * 16 + col;
#pragma unroll
        for (int v = 0; v < 8; ++v) {
            const float aa = accA[v] + ba_s;
            const float gg = accG[v] + bg_s;
            const float s  = 1.0f / (1.0f + __expf(-gg));
            yrow[(size_t)v * DM] = aa * s;
        }
    }
}

// ---------------------------------------------------------------------------
// launch
// ---------------------------------------------------------------------------
extern "C" void kernel_launch(void* const* d_in, const int* in_sizes, int n_in,
                              void* d_out, int out_size, void* d_ws, size_t ws_size,
                              hipStream_t stream) {
    const float*         x        = (const float*)d_in[0];
    const float*         features = (const float*)d_in[1];
    const unsigned char* mask     = (const unsigned char*)d_in[2];  // jnp bool = 1 byte
    const float*         W        = (const float*)d_in[3];
    const float*         bvec     = (const float*)d_in[4];
    float*               y        = (float*)d_out;

    char* ws = (char*)d_ws;
    float*    xn   = (float*)(ws);                       //  8 MB  [B,N,32]
    float*    xcrd = (float*)(ws + (8u  << 20));         //  4 MB  [B,N,16]
    int*      idx  = (int*)  (ws + (12u << 20));         //  4 MB  [B,N,16]
    unsigned* Wh   = (unsigned*)(ws + (16u << 20));      //  1 MB  packed f16 B-frags

    k_normalize<<<BB, 256, 0, stream>>>(x, features, mask, xn, xcrd);
    k_packW   <<<(64 * 16 * 32 * 8) / 256, 256, 0, stream>>>(W, Wh);
    k_knn     <<<(BB * NN) / 8, 256, 0, stream>>>(xcrd, mask, idx);
    k_gemm_glu<<<(BB * NN / 16) / 8, 256, 65536, stream>>>(xn, idx, Wh, bvec, y);
}
